// FusedMoE_30657476559669
// MI455X (gfx1250) — matrix operand order repaired and checked by reading, not compiled, not run on hardware
//
#include <hip/hip_runtime.h>
#include <stdint.h>

// Problem constants (match reference).
#define TT 8192
#define HH 2048
#define II 7168
#define EE 8

typedef __attribute__((ext_vector_type(16))) __bf16 v16bf;
typedef __attribute__((ext_vector_type(8)))  float  v8f;
typedef __attribute__((ext_vector_type(4)))  int    v4i_t;

struct Frag32B { uint4 lo, hi; };

// Load a 32-byte WMMA fragment (16 bf16) as two 16B LDS reads.
__device__ __forceinline__ v16bf ld_frag(const unsigned short* p0,
                                         const unsigned short* p1) {
  Frag32B f;
  f.lo = *(const uint4*)p0;
  f.hi = *(const uint4*)p1;
  return __builtin_bit_cast(v16bf, f);
}

// f32 -> bf16 round-to-nearest-even.
__device__ __forceinline__ unsigned short f2bf(float x) {
  unsigned int u = __builtin_bit_cast(unsigned int, x);
  u += 0x7FFFu + ((u >> 16) & 1u);
  return (unsigned short)(u >> 16);
}

__device__ __forceinline__ uint4 pack8(const float* f) {
  uint4 r;
  r.x = (unsigned)f2bf(f[0]) | ((unsigned)f2bf(f[1]) << 16);
  r.y = (unsigned)f2bf(f[2]) | ((unsigned)f2bf(f[3]) << 16);
  r.z = (unsigned)f2bf(f[4]) | ((unsigned)f2bf(f[5]) << 16);
  r.w = (unsigned)f2bf(f[6]) | ((unsigned)f2bf(f[7]) << 16);
  return r;
}

__device__ __forceinline__ void ld16f(const float* p, float* f) {
  float4 a = *(const float4*)(p + 0);
  float4 b = *(const float4*)(p + 4);
  float4 c = *(const float4*)(p + 8);
  float4 d = *(const float4*)(p + 12);
  f[0]=a.x; f[1]=a.y; f[2]=a.z; f[3]=a.w;
  f[4]=b.x; f[5]=b.y; f[6]=b.z; f[7]=b.w;
  f[8]=c.x; f[9]=c.y; f[10]=c.z; f[11]=c.w;
  f[12]=d.x; f[13]=d.y; f[14]=d.z; f[15]=d.w;
}

// gfx1250 async global->LDS copy (16 bytes per lane, tracked by ASYNCcnt).
// Builtin parameter types per hipcc diagnostic: (v4i __device__*, v4i __shared__*, imm, imm).
__device__ __forceinline__ void async_copy_b128(const unsigned short* gsrc,
                                                unsigned short* ldst) {
  __builtin_amdgcn_global_load_async_to_lds_b128(
      (__attribute__((address_space(1))) v4i_t*)(void*)gsrc,
      (__attribute__((address_space(3))) v4i_t*)(void*)ldst,
      /*imm offset=*/0, /*cpol=*/0);
}

__device__ __forceinline__ void wait_asynccnt0() {
  asm volatile("s_wait_asynccnt 0x0" ::: "memory");
}

// ---------------------------------------------------------------------------
// Kernel 0: zero output + expert counts (harness poisons buffers with 0xAA).
// ---------------------------------------------------------------------------
__global__ void moe_zero(float* __restrict__ out, int* __restrict__ counts) {
  size_t i = (size_t)blockIdx.x * blockDim.x + threadIdx.x;
  const size_t n = (size_t)TT * HH;
  const size_t step = (size_t)gridDim.x * blockDim.x;
  for (; i < n; i += step) out[i] = 0.0f;
  if (blockIdx.x == 0 && threadIdx.x < EE) counts[threadIdx.x] = 0;
}

// ---------------------------------------------------------------------------
// Kernel 1: routing. Top-2 of softmax(E=8), renormalized over the two picks.
//   w0 = e^{l0} / (e^{l0}+e^{l1}) = 1/(1+e^{l1-l0}),  w1 = 1-w0.
// ---------------------------------------------------------------------------
__global__ __launch_bounds__(256)
void moe_route(const float* __restrict__ logits,
               int* __restrict__ tch, float* __restrict__ wts) {
  const int t = blockIdx.x * 256 + threadIdx.x;
  if (t >= TT) return;
  float l[EE];
#pragma unroll
  for (int e = 0; e < EE; ++e) l[e] = logits[(size_t)t * EE + e];
  int i0 = 0; float v0 = l[0];
#pragma unroll
  for (int e = 1; e < EE; ++e) if (l[e] > v0) { v0 = l[e]; i0 = e; }
  int i1 = -1; float v1 = -3.0e38f;
#pragma unroll
  for (int e = 0; e < EE; ++e) if (e != i0 && l[e] > v1) { v1 = l[e]; i1 = e; }
  const float w0 = 1.0f / (1.0f + __expf(v1 - v0));
  wts[2 * t + 0] = w0;
  wts[2 * t + 1] = 1.0f - w0;
  tch[2 * t + 0] = i0;
  tch[2 * t + 1] = i1;
}

// ---------------------------------------------------------------------------
// Kernel 2: deterministic per-expert bucketing (wave w handles expert w).
// Token order preserved via ballot+popcount prefix; no atomics.
// Entry = 2*token + choice.
// ---------------------------------------------------------------------------
__global__ __launch_bounds__(256)
void moe_bucket(const int* __restrict__ tch,
                int* __restrict__ counts, int* __restrict__ bucket) {
  const int e = threadIdx.x >> 5;   // expert id
  const int lane = threadIdx.x & 31;
  int base = 0;
  for (int t0 = 0; t0 < TT; t0 += 32) {
    const int t = t0 + lane;
    const int c0 = tch[2 * t + 0];
    const int c1 = tch[2 * t + 1];
    const unsigned long long below = (1ull << lane) - 1ull;
    const unsigned long long m0 = __ballot(c0 == e);
    if (c0 == e) bucket[(size_t)e * TT + base + __popcll(m0 & below)] = 2 * t;
    base += __popcll(m0);
    const unsigned long long m1 = __ballot(c1 == e);
    if (c1 == e) bucket[(size_t)e * TT + base + __popcll(m1 & below)] = 2 * t + 1;
    base += __popcll(m1);
  }
  if (lane == 0) counts[e] = base;
}

// ---------------------------------------------------------------------------
// Kernel 3: gate-up GEMM + SiLU.  Block = 256 threads (8 waves).
// Tile: 128 tokens (M) x 128 intermediate cols (N), for both gate and up.
// K loop over H in steps of 32; f32 weights/activations converted to bf16
// on the way into LDS; 16 v_wmma_f32_16x16x32_bf16 per iteration per wave.
// h = silu(g)*u written as bf16 to the workspace intermediate buffer.
// ---------------------------------------------------------------------------
#define LDSTRIDE 40  // 32 + 8 pad: keeps 16B row alignment, spreads banks

__global__ __launch_bounds__(256)
void moe_gateup(const float* __restrict__ x, const float* __restrict__ w13,
                const int* __restrict__ counts, const int* __restrict__ bucket,
                unsigned short* __restrict__ hbuf) {
  const int e = blockIdx.z;
  const int cnt = counts[e];
  const int m0 = blockIdx.y * 128;
  if (m0 >= cnt) return;
  const int n0 = blockIdx.x * 128;

  __shared__ __align__(16) unsigned short xs[128 * LDSTRIDE];
  __shared__ __align__(16) unsigned short gs[128 * LDSTRIDE];
  __shared__ __align__(16) unsigned short us[128 * LDSTRIDE];
  __shared__ int prs[128];

  const int tid = threadIdx.x;
  const int lane = tid & 31;
  const int wv = tid >> 5;

  if (tid < 128) {
    const int idx = m0 + tid;
    prs[tid] = (idx < cnt) ? bucket[(size_t)e * TT + idx] : -1;
  }
  __syncthreads();

  const float* w13e = w13 + (size_t)e * (2 * (size_t)II) * HH;

  // Cooperative load plan: thread = (row 0..127, half 0..1); 16 f32 each.
  const int row  = tid >> 1;
  const int half = tid & 1;
  const int soff = row * LDSTRIDE + half * 16;
  const int pm   = prs[row];
  const float* xrow = (pm >= 0) ? (x + (size_t)(pm >> 1) * HH + half * 16) : nullptr;
  const float* grow = w13e + (size_t)(n0 + row) * HH + half * 16;
  const float* urow = w13e + ((size_t)II + n0 + row) * HH + half * 16;

  // Wave tile: wm in 0..3 (32 rows), wn in 0..1 (64 cols).
  const int wm = wv >> 1;
  const int wn = wv & 1;
  const int ksel = lane >> 4;
  const int lan15 = lane & 15;

  v8f cg[2][4], cu[2][4];
#pragma unroll
  for (int mi = 0; mi < 2; ++mi)
#pragma unroll
    for (int nj = 0; nj < 4; ++nj) { cg[mi][nj] = (v8f){}; cu[mi][nj] = (v8f){}; }

#pragma unroll 1
  for (int kk = 0; kk < HH; kk += 32) {
    float fx[16], fg[16], fu[16];
    if (xrow) ld16f(xrow + kk, fx);
    else { for (int j = 0; j < 16; ++j) fx[j] = 0.0f; }
    ld16f(grow + kk, fg);
    ld16f(urow + kk, fu);
    if (kk + 32 < HH) {  // warm L2/WGP$ for next weight K-tile
      __builtin_prefetch(grow + kk + 32, 0, 1);
      __builtin_prefetch(urow + kk + 32, 0, 1);
    }
    *(uint4*)&xs[soff + 0] = pack8(fx);
    *(uint4*)&xs[soff + 8] = pack8(fx + 8);
    *(uint4*)&gs[soff + 0] = pack8(fg);
    *(uint4*)&gs[soff + 8] = pack8(fg + 8);
    *(uint4*)&us[soff + 0] = pack8(fu);
    *(uint4*)&us[soff + 8] = pack8(fu + 8);
    __syncthreads();

#pragma unroll
    for (int mi = 0; mi < 2; ++mi) {
      const int ar = (wm * 32 + mi * 16 + lan15) * LDSTRIDE;
      const v16bf a = ld_frag(&xs[ar + ksel * 16], &xs[ar + ksel * 16 + 8]);
#pragma unroll
      for (int nj = 0; nj < 4; ++nj) {
        const int br = (wn * 64 + nj * 16 + lan15) * LDSTRIDE + ksel * 16;
        const v16bf bg = ld_frag(&gs[br], &gs[br + 8]);
        cg[mi][nj] = __builtin_amdgcn_wmma_f32_16x16x32_bf16(
            false, a, false, bg, (short)0, cg[mi][nj], false, false);
        const v16bf bu = ld_frag(&us[br], &us[br + 8]);
        cu[mi][nj] = __builtin_amdgcn_wmma_f32_16x16x32_bf16(
            false, a, false, bu, (short)0, cu[mi][nj], false, false);
      }
    }
    __syncthreads();
  }

  // Epilogue: h = silu(g) * u -> bf16 into hbuf[(2t+c)][I].
#pragma unroll
  for (int mi = 0; mi < 2; ++mi) {
#pragma unroll
    for (int nj = 0; nj < 4; ++nj) {
      const int mbase = wm * 32 + mi * 16 + ksel * 8;
      const int ncol = n0 + wn * 64 + nj * 16 + lan15;
#pragma unroll
      for (int i = 0; i < 8; ++i) {
        const int mloc = mbase + i;
        if (m0 + mloc < cnt) {
          const int p = prs[mloc];
          const float g = cg[mi][nj][i];
          const float u = cu[mi][nj][i];
          const float h = (g / (1.0f + __expf(-g))) * u;
          hbuf[(size_t)p * II + ncol] = f2bf(h);
        }
      }
    }
  }
}

// ---------------------------------------------------------------------------
// Kernel 4: down GEMM + weighted combine. Tile 128 pairs x 128 of H,
// K loop over I=7168 in steps of 32. The h tile is already bf16, so it is
// staged with gfx1250 async global->LDS b128 copies (ASYNCcnt); padding rows
// point at hbuf[0] — WMMA rows are independent and padded C rows are never
// stored, so their contents are don't-care. Atomic f32 combine.
// ---------------------------------------------------------------------------
__global__ __launch_bounds__(256)
void moe_down(const unsigned short* __restrict__ hbuf,
              const float* __restrict__ w2,
              const int* __restrict__ counts, const int* __restrict__ bucket,
              const float* __restrict__ wts, float* __restrict__ out) {
  const int e = blockIdx.z;
  const int cnt = counts[e];
  const int m0 = blockIdx.y * 128;
  if (m0 >= cnt) return;
  const int n0 = blockIdx.x * 128;

  __shared__ __align__(16) unsigned short as[128 * LDSTRIDE];
  __shared__ __align__(16) unsigned short bs[128 * LDSTRIDE];
  __shared__ int prs[128];
  __shared__ float wls[128];

  const int tid = threadIdx.x;
  const int lane = tid & 31;
  const int wv = tid >> 5;

  if (tid < 128) {
    const int idx = m0 + tid;
    const int p = (idx < cnt) ? bucket[(size_t)e * TT + idx] : -1;
    prs[tid] = p;
    wls[tid] = (p >= 0) ? wts[p] : 0.0f;
  }
  __syncthreads();

  const float* w2e = w2 + (size_t)e * (size_t)HH * II;

  const int row  = tid >> 1;
  const int half = tid & 1;
  const int soff = row * LDSTRIDE + half * 16;
  const int pm   = prs[row];
  // Padding rows read hbuf[0..]: harmless (see kernel comment).
  const unsigned short* hrow =
      hbuf + (pm >= 0 ? (size_t)pm * II : (size_t)0) + half * 16;
  const float* brow = w2e + (size_t)(n0 + row) * II + half * 16;

  const int wm = wv >> 1;
  const int wn = wv & 1;
  const int ksel = lane >> 4;
  const int lan15 = lane & 15;

  v8f c[2][4];
#pragma unroll
  for (int mi = 0; mi < 2; ++mi)
#pragma unroll
    for (int nj = 0; nj < 4; ++nj) c[mi][nj] = (v8f){};

#pragma unroll 1
  for (int kk = 0; kk < II; kk += 32) {
    // A tile: async byte copy, no VGPR round trip.
    async_copy_b128(hrow + kk + 0, &as[soff + 0]);
    async_copy_b128(hrow + kk + 8, &as[soff + 8]);

    // B tile: f32 weights -> bf16 in registers -> LDS.
    float fb[16];
    ld16f(brow + kk, fb);
    if (kk + 32 < II) __builtin_prefetch(brow + kk + 32, 0, 1);
    *(uint4*)&bs[soff + 0] = pack8(fb);
    *(uint4*)&bs[soff + 8] = pack8(fb + 8);

    wait_asynccnt0();
    __syncthreads();

#pragma unroll
    for (int mi = 0; mi < 2; ++mi) {
      const int ar = (wm * 32 + mi * 16 + lan15) * LDSTRIDE;
      const v16bf a = ld_frag(&as[ar + ksel * 16], &as[ar + ksel * 16 + 8]);
#pragma unroll
      for (int nj = 0; nj < 4; ++nj) {
        const int br = (wn * 64 + nj * 16 + lan15) * LDSTRIDE + ksel * 16;
        const v16bf b = ld_frag(&bs[br], &bs[br + 8]);
        c[mi][nj] = __builtin_amdgcn_wmma_f32_16x16x32_bf16(
            false, a, false, b, (short)0, c[mi][nj], false, false);
      }
    }
    __syncthreads();
  }

  // Epilogue: out[t][n] += route_weight * c
#pragma unroll
  for (int mi = 0; mi < 2; ++mi) {
#pragma unroll
    for (int nj = 0; nj < 4; ++nj) {
      const int mbase = wm * 32 + mi * 16 + ksel * 8;
      const int ncol = n0 + wn * 64 + nj * 16 + lan15;
#pragma unroll
      for (int i = 0; i < 8; ++i) {
        const int mloc = mbase + i;
        if (m0 + mloc < cnt) {
          const int p = prs[mloc];
          const int t = p >> 1;
          atomicAdd(out + (size_t)t * HH + ncol, c[mi][nj][i] * wls[mloc]);
        }
      }
    }
  }
}

// ---------------------------------------------------------------------------
// Launch. Workspace layout (ints/floats/bf16), total ~235.4 MB:
//   counts[16] | bucket[E*T] | tch[2T] | wts[2T] | hbuf[2T*I] (bf16)
// ---------------------------------------------------------------------------
extern "C" void kernel_launch(void* const* d_in, const int* in_sizes, int n_in,
                              void* d_out, int out_size, void* d_ws, size_t ws_size,
                              hipStream_t stream) {
  (void)in_sizes; (void)n_in; (void)out_size; (void)ws_size;
  const float* x   = (const float*)d_in[0];
  const float* rl  = (const float*)d_in[1];
  const float* w13 = (const float*)d_in[2];
  const float* w2  = (const float*)d_in[3];
  float* out = (float*)d_out;

  int* counts = (int*)d_ws;
  int* bucket = counts + 16;
  int* tch    = bucket + (size_t)EE * TT;
  float* wts  = (float*)(tch + 2 * TT);
  unsigned short* hbuf = (unsigned short*)(wts + 2 * TT);

  moe_zero<<<1024, 256, 0, stream>>>(out, counts);
  moe_route<<<TT / 256, 256, 0, stream>>>(rl, tch, wts);
  moe_bucket<<<1, 256, 0, stream>>>(tch, counts, bucket);

  dim3 gA(II / 128, TT / 128, EE);
  moe_gateup<<<gA, 256, 0, stream>>>(x, w13, counts, bucket, hbuf);

  dim3 gB(HH / 128, TT / 128, EE);
  moe_down<<<gB, 256, 0, stream>>>(hbuf, w2, counts, bucket, wts, out);
}
